// HandNet_19902878450320
// MI455X (gfx1250) — compile-verified, matrix operand order checked by reading, not run on hardware
//
#include <hip/hip_runtime.h>
#include <cstddef>
#include <cstdint>

// ---------------- CDNA5 WMMA vector types ----------------
typedef __attribute__((ext_vector_type(16))) __bf16 v16bf;
typedef __attribute__((ext_vector_type(8)))  __bf16 v8bf;
typedef __attribute__((ext_vector_type(8)))  float  v8f;

// ---------------- problem constants ----------------
#define BGR   8192
#define JS    17
#define ES    16
#define JT    18
#define ET    17
#define NSRC  (2*BGR*JS)          // 278528 encoder nodes
#define NTGT  (2*BGR*JT)          // 294912 decoder nodes
#define HNODE (BGR*JT)            // 147456
#define TM    16384               // trans GEMM M (= 2*B graphs)
#define TK    1088                // 17*64
#define TN    1152                // 18*64

// ---------------- d_out layout (float elements) ----------------
#define OFF_Z    ((size_t)0)
#define OFF_ANG0 ((size_t)TM*(size_t)TN)          // 18,874,368
#define OFF_POS0 (OFF_ANG0 + (size_t)HNODE)
#define OFF_ANG1 (OFF_POS0 + (size_t)3*HNODE)
#define OFF_POS1 (OFF_ANG1 + (size_t)HNODE)

// ---------------- workspace layout (bytes) ----------------
#define WS_H1   ((size_t)0)                 // 278528*16*4 = 17,825,792
#define WS_H2   ((size_t)17825792)          // 278528*32*4 = 35,651,584
#define WS_H3BF ((size_t)53477376)          // 278528*64*2 = 35,651,584 (bf16 A)
#define WS_BT   ((size_t)89128960)          // 1152*1088*2 =  2,506,752 (bf16 W^T)
#define WS_HD1  WS_H1                       // reuse: 294912*32*4 fits in H1+H2
#define WS_HD2  WS_H3BF                     // reuse: 294912*16*4 fits in H3BF
#define WS_ANG  WS_BT                       // reuse: 294912*4 fits in BT
#define WS_NEED ((size_t)91635712)

// ---------------- helpers ----------------
__device__ __forceinline__ float getf(const float* __restrict__ a,
                                      const float* __restrict__ b,
                                      int halfRows, int row, int nf, int k) {
  if (b != nullptr && row >= halfRows)
    return b[(size_t)(row - halfRows) * nf + k];
  return a[(size_t)row * nf + k];
}

// =========================================================================
// Generic per-node GNN block. Chain topology: node j>=1 receives exactly one
// message from node j-1 through edge j-1 of its graph. Features staged in LDS.
// COUT must divide 256; nNodes must be a multiple of 256/COUT (true here).
// =========================================================================
template<int CIN, int ED, int COUT, int J, int EPER>
__global__ __launch_bounds__(256)
void gnn_block_kernel(const float* __restrict__ xa, const float* __restrict__ xb, int halfX,
                      const float* __restrict__ ea, const float* __restrict__ eb, int halfE,
                      const float* __restrict__ Wl, const float* __restrict__ bl,
                      const float* __restrict__ Wu, const float* __restrict__ bu,
                      float* __restrict__ out, __bf16* __restrict__ outbf, int nNodes) {
  constexpr int NPB  = 256 / COUT;
  constexpr int FTOT = 2 * CIN + ED;
  __shared__ float feat[NPB][FTOT];
  const int node0 = blockIdx.x * NPB;

  for (int e = threadIdx.x; e < NPB * FTOT; e += 256) {
    const int nl = e / FTOT, f = e % FTOT;
    const int node = node0 + nl;
    const int j = node % J;
    float v = 0.f;
    if (f < CIN) {
      v = getf(xa, xb, halfX, node, CIN, f);
    } else if (f < 2 * CIN) {
      if (j > 0) v = getf(xa, xb, halfX, node - 1, CIN, f - CIN);
    } else {
      if (j > 0) v = getf(ea, eb, halfE, (node / J) * EPER + (j - 1), ED, f - 2 * CIN);
    }
    feat[nl][f] = v;
  }
  __syncthreads();

  const int nl   = threadIdx.x / COUT;
  const int oc   = threadIdx.x % COUT;
  const int node = node0 + nl;
  if (node >= nNodes) return;
  const int j = node % J;
  const float* F = feat[nl];

  float accu = bu[oc];
  #pragma unroll 4
  for (int k = 0; k < CIN; ++k) accu += F[k] * Wu[k * COUT + oc];
  float res = accu;
  if (j > 0) {
    float accm = bl[oc];
    #pragma unroll 4
    for (int k = 0; k < FTOT; ++k) accm += F[k] * Wl[k * COUT + oc];
    res += (accm > 0.f) ? accm : 0.01f * accm;       // leaky_relu(0.01)
  }
  if (out)   out[(size_t)node * COUT + oc]   = res;
  if (outbf) outbf[(size_t)node * COUT + oc] = (__bf16)res;
}

// =========================================================================
// trans_w (K=1088 x N=1152, f32 row-major)  ->  Bt (N x K, bf16 row-major)
// =========================================================================
__global__ __launch_bounds__(256)
void wtrans_kernel(const float* __restrict__ W, __bf16* __restrict__ Bt) {
  const int t = blockIdx.x * 256 + threadIdx.x;
  if (t >= TK * TN) return;
  const int k = t / TN, n = t % TN;
  Bt[(size_t)n * TK + k] = (__bf16)W[t];
}

// =========================================================================
// trans GEMM: Z = tanh(A(16384x1088,bf16) * W(1088x1152) + b), via
// v_wmma_f32_16x16x32_bf16.
//
// Register tiling: each wave owns a 32(M) x 64(N) output tile -> 2 A frags,
// 4 B frags, 8 WMMA accumulators. Per 32-deep K step a lane issues 12 b128
// loads (6144B/wave, 192B/lane total across A+B) feeding 8 WMMAs
// (8*16384 = 131072 FLOP/wave-step) -> ~22 FLOP per L2 byte, vs ~8 for a
// single-tile wave. A is re-read 18x (not 72x), B 512x (not 1024x):
// ~1.9 GB total L2 traffic for the 41-GFLOP GEMM.
//
// CDNA5 16-bit fragment layout: lane L<16 holds row/col (base+L) with
// K = kb+{0..7,16..23}; lanes 16..31 hold K = kb+{8..15,24..31}. Both are
// two contiguous 16B loads from K-major storage (A row-major, Bt N-major).
// =========================================================================
#define MTW 32                    // M rows per wave tile
#define NTW 64                    // N cols per wave tile

__device__ __forceinline__ v16bf load_frag(const __bf16* __restrict__ p) {
  v16bf v;
  *((v8bf*)&v)     = *(const v8bf*)(p);        // K kb+0..7  (or +8..15)
  *((v8bf*)&v + 1) = *(const v8bf*)(p + 16);   // K kb+16..23 (or +24..31)
  return v;
}

__global__ __launch_bounds__(256)
void trans_wmma_kernel(const __bf16* __restrict__ A,
                       const __bf16* __restrict__ Bt,
                       const float* __restrict__ bias,
                       float* __restrict__ Z) {
  const int lane  = threadIdx.x & 31;
  const int wave  = threadIdx.x >> 5;
  const int tile  = blockIdx.x * 8 + wave;     // (16384/32)*(1152/64) = 9216
  const int nt    = TN / NTW;                  // 18
  const int m0    = (tile / nt) * MTW;
  const int n0    = (tile % nt) * NTW;
  const int l15   = lane & 15;
  const int khalf = (lane >> 4) ? 8 : 0;

  const __bf16* aRow0 = A  + (size_t)(m0 + l15) * TK + khalf;
  const __bf16* aRow1 = aRow0 + (size_t)16 * TK;
  const __bf16* bRow  = Bt + (size_t)(n0 + l15) * TK + khalf;
  const size_t  bStep = (size_t)16 * TK;       // 16 N-columns

  v8f c[2][4] = {};
  for (int k = 0; k < TK; k += 32) {
    v16bf a0 = load_frag(aRow0 + k);
    v16bf a1 = load_frag(aRow1 + k);
    v16bf b0 = load_frag(bRow + 0 * bStep + k);
    v16bf b1 = load_frag(bRow + 1 * bStep + k);
    v16bf b2 = load_frag(bRow + 2 * bStep + k);
    v16bf b3 = load_frag(bRow + 3 * bStep + k);
    __builtin_prefetch(aRow0 + k + 32, 0, 0);  // global_prefetch_b8
    __builtin_prefetch(aRow1 + k + 32, 0, 0);
    __builtin_prefetch(bRow + 0 * bStep + k + 32, 0, 0);
    __builtin_prefetch(bRow + 1 * bStep + k + 32, 0, 0);
    __builtin_prefetch(bRow + 2 * bStep + k + 32, 0, 0);
    __builtin_prefetch(bRow + 3 * bStep + k + 32, 0, 0);
    c[0][0] = __builtin_amdgcn_wmma_f32_16x16x32_bf16(false, a0, false, b0, (short)0, c[0][0], false, false);
    c[0][1] = __builtin_amdgcn_wmma_f32_16x16x32_bf16(false, a0, false, b1, (short)0, c[0][1], false, false);
    c[0][2] = __builtin_amdgcn_wmma_f32_16x16x32_bf16(false, a0, false, b2, (short)0, c[0][2], false, false);
    c[0][3] = __builtin_amdgcn_wmma_f32_16x16x32_bf16(false, a0, false, b3, (short)0, c[0][3], false, false);
    c[1][0] = __builtin_amdgcn_wmma_f32_16x16x32_bf16(false, a1, false, b0, (short)0, c[1][0], false, false);
    c[1][1] = __builtin_amdgcn_wmma_f32_16x16x32_bf16(false, a1, false, b1, (short)0, c[1][1], false, false);
    c[1][2] = __builtin_amdgcn_wmma_f32_16x16x32_bf16(false, a1, false, b2, (short)0, c[1][2], false, false);
    c[1][3] = __builtin_amdgcn_wmma_f32_16x16x32_bf16(false, a1, false, b3, (short)0, c[1][3], false, false);
  }

  // D layout per 16x16 tile: VGPR r, lanes 0-15 -> (M=base+r, N=n+lane),
  //                                   lanes16-31 -> (M=base+8+r, N=n+lane-16)
  #pragma unroll
  for (int ni = 0; ni < 4; ++ni) {
    const int n  = n0 + ni * 16 + l15;
    const float bn = bias[n];
    #pragma unroll
    for (int mi = 0; mi < 2; ++mi) {
      const int mBase = m0 + mi * 16 + khalf;
      #pragma unroll
      for (int r = 0; r < 8; ++r)
        Z[(size_t)(mBase + r) * TN + n] = tanhf(c[mi][ni][r] + bn);
    }
  }
}

// =========================================================================
// dec1: input features are [z(64) | lower(1) | upper(1)] per node (CIN=66).
// z lives in d_out (row-major 294912 x 64); lo/up are B*JT arrays, concat'd.
// =========================================================================
__device__ __forceinline__ float dec_feat(const float* __restrict__ z,
                                          const float* __restrict__ lo,
                                          const float* __restrict__ up,
                                          int row, int k) {
  if (k < 64) return z[(size_t)row * 64 + k];
  const int il = row % HNODE;
  return (k == 64) ? lo[il] : up[il];
}

__global__ __launch_bounds__(256)
void dec1_kernel(const float* __restrict__ z,
                 const float* __restrict__ lo, const float* __restrict__ up,
                 const float* __restrict__ ea,
                 const float* __restrict__ Wl, const float* __restrict__ bl,
                 const float* __restrict__ Wu, const float* __restrict__ bu,
                 float* __restrict__ out, int nNodes) {
  constexpr int CIN = 66, ED = 6, COUT = 32, J = JT, EPER = ET;
  constexpr int NPB = 256 / COUT;           // 8 nodes per block
  constexpr int FTOT = 2 * CIN + ED;        // 138
  __shared__ float feat[NPB][FTOT];
  const int node0 = blockIdx.x * NPB;
  const int halfE = BGR * EPER;             // 139264 (ea is self-concat)

  for (int e = threadIdx.x; e < NPB * FTOT; e += 256) {
    const int nl = e / FTOT, f = e % FTOT;
    const int node = node0 + nl;
    const int j = node % J;
    float v = 0.f;
    if (f < CIN) {
      v = dec_feat(z, lo, up, node, f);
    } else if (f < 2 * CIN) {
      if (j > 0) v = dec_feat(z, lo, up, node - 1, f - CIN);
    } else {
      if (j > 0) {
        const int er = ((node / J) * EPER + (j - 1)) % halfE;
        v = ea[(size_t)er * ED + (f - 2 * CIN)];
      }
    }
    feat[nl][f] = v;
  }
  __syncthreads();

  const int nl = threadIdx.x / COUT;
  const int oc = threadIdx.x % COUT;
  const int node = node0 + nl;
  if (node >= nNodes) return;
  const int j = node % J;
  const float* F = feat[nl];

  float accu = bu[oc];
  #pragma unroll 4
  for (int k = 0; k < CIN; ++k) accu += F[k] * Wu[k * COUT + oc];
  float res = accu;
  if (j > 0) {
    float accm = bl[oc];
    #pragma unroll 4
    for (int k = 0; k < FTOT; ++k) accm += F[k] * Wl[k * COUT + oc];
    res += (accm > 0.f) ? accm : 0.01f * accm;
  }
  out[(size_t)node * COUT + oc] = res;
}

// =========================================================================
// dec3 (16 -> 1) fused with tanh + joint-limit mapping -> ang, scattered
// straight into d_out's [ang0 | ... | ang1] segments and kept in ws for FK.
// =========================================================================
__global__ __launch_bounds__(256)
void dec3_ang_kernel(const float* __restrict__ h,
                     const float* __restrict__ ea,
                     const float* __restrict__ Wl, const float* __restrict__ bl,
                     const float* __restrict__ Wu, const float* __restrict__ bu,
                     const float* __restrict__ lo, const float* __restrict__ up,
                     float* __restrict__ ang_ws, float* __restrict__ outp, int nNodes) {
  const int node = blockIdx.x * 256 + threadIdx.x;
  if (node >= nNodes) return;
  const int j = node % JT, g = node / JT;

  float accu = bu[0];
  #pragma unroll
  for (int k = 0; k < 16; ++k) accu += h[(size_t)node * 16 + k] * Wu[k];
  float res = accu;
  if (j > 0) {
    float accm = bl[0];
    #pragma unroll
    for (int k = 0; k < 16; ++k) accm += h[(size_t)node * 16 + k] * Wl[k];
    #pragma unroll
    for (int k = 0; k < 16; ++k) accm += h[(size_t)(node - 1) * 16 + k] * Wl[16 + k];
    const int er = (g * ET + (j - 1)) % (BGR * ET);
    #pragma unroll
    for (int k = 0; k < 6; ++k) accm += ea[(size_t)er * 6 + k] * Wl[32 + k];
    res += (accm > 0.f) ? accm : 0.01f * accm;
  }
  const float hd = tanhf(res);
  const int il = node % HNODE;
  const float l = lo[il], u = up[il];
  const float a = l + (u - l) * (hd + 1.f) * 0.5f;
  ang_ws[node] = a;
  if (node < HNODE) outp[OFF_ANG0 + node] = a;
  else              outp[OFF_ANG1 + (node - HNODE)] = a;
}

// =========================================================================
// Forward kinematics: one thread per graph (16384), 18-joint Rodrigues chain.
// =========================================================================
__device__ __forceinline__ void mat3mul(const float* a, const float* b, float* c) {
  #pragma unroll
  for (int r = 0; r < 3; ++r)
    #pragma unroll
    for (int q = 0; q < 3; ++q)
      c[r * 3 + q] = a[r * 3 + 0] * b[0 * 3 + q]
                   + a[r * 3 + 1] * b[1 * 3 + q]
                   + a[r * 3 + 2] * b[2 * 3 + q];
}

__global__ __launch_bounds__(64)
void fk_kernel(const float* __restrict__ ang,
               const int*   __restrict__ parent,
               const float* __restrict__ offset,
               const float* __restrict__ axis,
               float* __restrict__ outp) {
  const int g = blockIdx.x * 64 + threadIdx.x;
  if (g >= TM) return;
  const int base  = g * JT;            // global node base
  const int ibase = (g % BGR) * JT;    // index into B-sized param arrays

  float rot[JT][9];
  float pos[JT][3];
  #pragma unroll
  for (int j = 0; j < JT; ++j) {
    #pragma unroll
    for (int e = 0; e < 9; ++e) rot[j][e] = 0.f;
    pos[j][0] = pos[j][1] = pos[j][2] = 0.f;
  }

  for (int j = 0; j < JT; ++j) {
    const float a  = ang[base + j];
    const float ax = axis[(size_t)(ibase + j) * 3 + 0];
    const float ay = axis[(size_t)(ibase + j) * 3 + 1];
    const float az = axis[(size_t)(ibase + j) * 3 + 2];
    const float s = sinf(a), c = cosf(a), oc = 1.f - c;

    const float K[9] = { 0.f, -az,  ay,
                         az,  0.f, -ax,
                        -ay,  ax,  0.f };
    float K2[9];
    mat3mul(K, K, K2);
    float R[9];
    #pragma unroll
    for (int e = 0; e < 9; ++e) R[e] = s * K[e] + oc * K2[e];
    R[0] += 1.f; R[4] += 1.f; R[8] += 1.f;

    const int p = parent[ibase + j];
    const float ox = offset[(size_t)(ibase + j) * 3 + 0];
    const float oy = offset[(size_t)(ibase + j) * 3 + 1];
    const float oz = offset[(size_t)(ibase + j) * 3 + 2];

    if (p < 0) {
      #pragma unroll
      for (int e = 0; e < 9; ++e) rot[j][e] = R[e];
      pos[j][0] = ox; pos[j][1] = oy; pos[j][2] = oz;
    } else {
      int ps = p; if (ps > JT - 1) ps = JT - 1;
      mat3mul(rot[ps], R, rot[j]);
      pos[j][0] = pos[ps][0] + rot[ps][0] * ox + rot[ps][1] * oy + rot[ps][2] * oz;
      pos[j][1] = pos[ps][1] + rot[ps][3] * ox + rot[ps][4] * oy + rot[ps][5] * oz;
      pos[j][2] = pos[ps][2] + rot[ps][6] * ox + rot[ps][7] * oy + rot[ps][8] * oz;
    }
  }

  for (int j = 0; j < JT; ++j) {
    const int node = base + j;
    size_t o = (node < HNODE) ? (OFF_POS0 + (size_t)3 * node)
                              : (OFF_POS1 + (size_t)3 * (node - HNODE));
    outp[o + 0] = pos[j][0];
    outp[o + 1] = pos[j][1];
    outp[o + 2] = pos[j][2];
  }
}

// =========================================================================
// Host-side launch
// =========================================================================
extern "C" void kernel_launch(void* const* d_in, const int* in_sizes, int n_in,
                              void* d_out, int out_size, void* d_ws, size_t ws_size,
                              hipStream_t stream) {
  (void)in_sizes; (void)n_in; (void)out_size;
  if (ws_size < WS_NEED) return;

  const float* l_x    = (const float*)d_in[0];
  const float* r_x    = (const float*)d_in[1];
  const float* l_ea   = (const float*)d_in[4];
  const float* r_ea   = (const float*)d_in[5];
  const float* t_ea   = (const float*)d_in[7];
  const float* lo     = (const float*)d_in[8];
  const float* up     = (const float*)d_in[9];
  const float* offp   = (const float*)d_in[10];
  const int*   parent = (const int*)  d_in[11];
  const float* axisp  = (const float*)d_in[12];

  const float* e1lw = (const float*)d_in[14]; const float* e1lb = (const float*)d_in[15];
  const float* e1uw = (const float*)d_in[16]; const float* e1ub = (const float*)d_in[17];
  const float* e2lw = (const float*)d_in[18]; const float* e2lb = (const float*)d_in[19];
  const float* e2uw = (const float*)d_in[20]; const float* e2ub = (const float*)d_in[21];
  const float* e3lw = (const float*)d_in[22]; const float* e3lb = (const float*)d_in[23];
  const float* e3uw = (const float*)d_in[24]; const float* e3ub = (const float*)d_in[25];
  const float* d1lw = (const float*)d_in[26]; const float* d1lb = (const float*)d_in[27];
  const float* d1uw = (const float*)d_in[28]; const float* d1ub = (const float*)d_in[29];
  const float* d2lw = (const float*)d_in[30]; const float* d2lb = (const float*)d_in[31];
  const float* d2uw = (const float*)d_in[32]; const float* d2ub = (const float*)d_in[33];
  const float* d3lw = (const float*)d_in[34]; const float* d3lb = (const float*)d_in[35];
  const float* d3uw = (const float*)d_in[36]; const float* d3ub = (const float*)d_in[37];
  const float* trW  = (const float*)d_in[38]; const float* trB  = (const float*)d_in[39];

  uint8_t* ws = (uint8_t*)d_ws;
  float*  H1  = (float*) (ws + WS_H1);
  float*  H2  = (float*) (ws + WS_H2);
  __bf16* H3  = (__bf16*)(ws + WS_H3BF);
  __bf16* BT  = (__bf16*)(ws + WS_BT);
  float*  HD1 = (float*) (ws + WS_HD1);
  float*  HD2 = (float*) (ws + WS_HD2);
  float*  ANG = (float*) (ws + WS_ANG);
  float*  out = (float*)d_out;
  float*  Z   = out + OFF_Z;

  const int halfX = BGR * JS;   // 139264 rows in each of l_x / r_x
  const int halfE = BGR * ES;   // 131072 rows in each of l_ea / r_ea

  // W^T -> bf16 (independent of encoder chain, launch first)
  wtrans_kernel<<<(TK * TN + 255) / 256, 256, 0, stream>>>(trW, BT);

  // encoder: 3 -> 16 -> 32 -> 64 (enc3 emits bf16 GEMM operand A)
  gnn_block_kernel<3, 3, 16, JS, ES><<<NSRC / 16, 256, 0, stream>>>(
      l_x, r_x, halfX, l_ea, r_ea, halfE,
      e1lw, e1lb, e1uw, e1ub, H1, nullptr, NSRC);
  gnn_block_kernel<16, 3, 32, JS, ES><<<NSRC / 8, 256, 0, stream>>>(
      H1, nullptr, 0, l_ea, r_ea, halfE,
      e2lw, e2lb, e2uw, e2ub, H2, nullptr, NSRC);
  gnn_block_kernel<32, 3, 64, JS, ES><<<NSRC / 4, 256, 0, stream>>>(
      H2, nullptr, 0, l_ea, r_ea, halfE,
      e3lw, e3lb, e3uw, e3ub, nullptr, H3, NSRC);

  // z = tanh(h @ trans_w + trans_b) — WMMA bf16, 32x64 wave tiles, into d_out
  trans_wmma_kernel<<<(TM / MTW) * (TN / NTW) / 8, 256, 0, stream>>>(H3, BT, trB, Z);

  // decoder: 66 -> 32 -> 16 -> 1 (+ tanh + joint limits), then FK
  dec1_kernel<<<NTGT / 8, 256, 0, stream>>>(
      Z, lo, up, t_ea, d1lw, d1lb, d1uw, d1ub, HD1, NTGT);
  gnn_block_kernel<32, 6, 16, JT, ET><<<NTGT / 16, 256, 0, stream>>>(
      HD1, nullptr, 0, t_ea, t_ea, BGR * ET,
      d2lw, d2lb, d2uw, d2ub, HD2, nullptr, NTGT);
  dec3_ang_kernel<<<NTGT / 256, 256, 0, stream>>>(
      HD2, t_ea, d3lw, d3lb, d3uw, d3ub, lo, up, ANG, out, NTGT);
  fk_kernel<<<TM / 64, 64, 0, stream>>>(ANG, parent, offp, axisp, out);
}
